// EncoderLayer_65128884077226
// MI455X (gfx1250) — compile-verified, matrix operand order checked by reading, not run on hardware
//
#include <hip/hip_runtime.h>
#include <hip/hip_bf16.h>
#include <math.h>

// ---------------- problem dims ----------------
#define BB    4
#define SS    1024
#define EE    512
#define FFD   2048
#define MROWS (BB * SS)      // 4096
#define LNEPS 1e-5f

typedef __attribute__((ext_vector_type(16))) _Float16 v16h;
typedef __attribute__((ext_vector_type(8)))  _Float16 v8h;
typedef __attribute__((ext_vector_type(8)))  float    v8f;

// ---- CDNA5 async global->LDS path (guarded so missing builtins fall back) ----
#if defined(__AMDGCN__) && __has_builtin(__builtin_amdgcn_global_load_async_to_lds_b128)
#define USE_ASYNC 1
// exact param types per hipcc diagnostic: (AS1 int4*, AS3 int4*, imm, imm)
typedef int v4i_t __attribute__((vector_size(16)));
typedef __attribute__((address_space(1))) v4i_t* as1_v4i_p;
typedef __attribute__((address_space(3))) v4i_t* as3_v4i_p;
#define ASYNC_CP16(gp, lp, off)                                              \
    __builtin_amdgcn_global_load_async_to_lds_b128(                          \
        (as1_v4i_p)(gp), (as3_v4i_p)(lp), (off), 0)
#if __has_builtin(__builtin_amdgcn_s_wait_asynccnt)
#define WAIT_ASYNC(n) __builtin_amdgcn_s_wait_asynccnt(n)
#else
#define WAIT_ASYNC(n) asm volatile("s_wait_asynccnt %0" ::"i"(n) : "memory")
#endif
#else
#define USE_ASYNC 0
#endif

// ---------------- block reduction ----------------
__device__ __forceinline__ float blockReduceSum(float v, float* smem) {
    #pragma unroll
    for (int off = 16; off > 0; off >>= 1) v += __shfl_xor(v, off, 32);
    const int lane = threadIdx.x & 31;
    const int w    = threadIdx.x >> 5;
    if (lane == 0) smem[w] = v;
    __syncthreads();
    const int nw = blockDim.x >> 5;
    v = (threadIdx.x < (unsigned)nw) ? smem[threadIdx.x] : 0.0f;
    if (w == 0) {
        #pragma unroll
        for (int off = 4; off > 0; off >>= 1) v += __shfl_xor(v, off, 32);
        if (lane == 0) smem[0] = v;
    }
    __syncthreads();
    float r = smem[0];
    __syncthreads();
    return r;
}

// ---------------- K1: LN1 row stats ----------------
__global__ __launch_bounds__(256) void k_ln_stats(const float* __restrict__ x,
                                                  float* __restrict__ mu,
                                                  float* __restrict__ rstd) {
    const int row = blockIdx.x;
    const float* xr = x + (size_t)row * EE;
    float s = 0.f, ss = 0.f;
    for (int e = threadIdx.x; e < EE; e += blockDim.x) {
        float v = xr[e];
        s += v; ss += v * v;
    }
    __shared__ float sm[8];
    s  = blockReduceSum(s,  sm);
    ss = blockReduceSum(ss, sm);
    if (threadIdx.x == 0) {
        float m   = s * (1.0f / EE);
        float var = ss * (1.0f / EE) - m * m;
        mu[row]   = m;
        rstd[row] = rsqrtf(var + LNEPS);
    }
}

// ---------------- K2: column mean of LN1(x) over sequence ----------------
__global__ __launch_bounds__(256) void k_colmean(const float* __restrict__ x,
                                                 const float* __restrict__ mu,
                                                 const float* __restrict__ rstd,
                                                 const float* __restrict__ w,
                                                 const float* __restrict__ bvec,
                                                 float* __restrict__ colmean) {
    const int e = blockIdx.x * blockDim.x + threadIdx.x;   // 0..511
    const int b = blockIdx.y;
    const float* xb = x    + (size_t)b * SS * EE;
    const float* mb = mu   + (size_t)b * SS;
    const float* rb = rstd + (size_t)b * SS;
    float acc = 0.f;
    for (int s = 0; s < SS; ++s)
        acc += (xb[(size_t)s * EE + e] - mb[s]) * rb[s];
    colmean[b * EE + e] = w[e] * (acc * (1.0f / SS)) + bvec[e];
}

// ---------------- K3: tiny per-batch GEMV ----------------
__global__ __launch_bounds__(256) void k_gemv(const float* __restrict__ W,
                                              const float* __restrict__ bias,
                                              const float* __restrict__ vin,
                                              float* __restrict__ vout) {
    __shared__ float vs[EE];
    const int b = blockIdx.y;
    for (int k = threadIdx.x; k < EE; k += blockDim.x) vs[k] = vin[b * EE + k];
    __syncthreads();
    const int n = blockIdx.x * blockDim.x + threadIdx.x;   // 0..511
    const float* wr = W + (size_t)n * EE;
    float acc = bias[n];
    for (int k = 0; k < EE; ++k) acc += wr[k] * vs[k];
    vout[b * EE + n] = acc;
}

// ---------------- K4: x2 = x + attnvec; LN2 -> f16 ----------------
__global__ __launch_bounds__(256) void k_add_ln2(const float* __restrict__ x,
                                                 const float* __restrict__ attnvec,
                                                 const float* __restrict__ w,
                                                 const float* __restrict__ bvec,
                                                 float* __restrict__ x2,
                                                 _Float16* __restrict__ hln) {
    const int row = blockIdx.x;            // 0..4095
    const int b   = row >> 10;             // S = 1024
    const float* xr = x + (size_t)row * EE;
    const float* av = attnvec + (size_t)b * EE;
    float loc[2];
    float s = 0.f, ss = 0.f;
    #pragma unroll
    for (int i = 0; i < 2; ++i) {
        const int e = threadIdx.x + i * 256;
        float v = xr[e] + av[e];
        loc[i] = v;
        x2[(size_t)row * EE + e] = v;
        s += v; ss += v * v;
    }
    __shared__ float sm[8];
    s  = blockReduceSum(s,  sm);
    ss = blockReduceSum(ss, sm);
    const float m = s * (1.0f / EE);
    const float r = rsqrtf(ss * (1.0f / EE) - m * m + LNEPS);
    #pragma unroll
    for (int i = 0; i < 2; ++i) {
        const int e = threadIdx.x + i * 256;
        hln[(size_t)row * EE + e] = (_Float16)((loc[i] - m) * r * w[e] + bvec[e]);
    }
}

// ---------------- K5: f32 -> f16 convert ----------------
__global__ __launch_bounds__(256) void k_f32_to_f16(const float* __restrict__ src,
                                                    _Float16* __restrict__ dst, int n) {
    const int i = blockIdx.x * blockDim.x + threadIdx.x;
    if (i < n) dst[i] = (_Float16)src[i];
}

// ---------------- WMMA GEMM:  C[M,N] = A[M,K] * B[N,K]^T + epilogue ----------------
// Block tile 128x128, 8 waves (2 x 4), wave tile 64x32 = 4x2 WMMA accumulators.
// Double-buffered LDS staged with GLOBAL_LOAD_ASYNC_TO_LDS_B128 (ASYNCcnt) when
// available; synchronous uint4 + ds_store fallback otherwise.
#define Bb 128
#define BN 128
#define BKK 32
#define PADH 8     // pad LDS row to 40 halves (80 B)

template <int EPI>  // 0 = bias+GELU -> f16, 1 = bias+residual -> f32
__global__ __launch_bounds__(256) void gemm_wmma(const _Float16* __restrict__ A,
                                                 const _Float16* __restrict__ Bw,
                                                 const float* __restrict__ bias,
                                                 const float* __restrict__ Res,
                                                 _Float16* __restrict__ outH,
                                                 float* __restrict__ outF,
                                                 int M, int N, int K) {
#if USE_ASYNC
    __shared__ _Float16 As[2][Bb][BKK + PADH];
    __shared__ _Float16 Bs[2][BN][BKK + PADH];
#else
    __shared__ _Float16 As[1][Bb][BKK + PADH];
    __shared__ _Float16 Bs[1][BN][BKK + PADH];
#endif

    const int tid  = threadIdx.x;
    const int lane = tid & 31;
    const int wid  = tid >> 5;     // 0..7
    const int wm   = wid & 1;      // 64-row slab
    const int wn   = wid >> 1;     // 32-col slab
    const int hi16 = lane >> 4;    // lane group 0/1
    const int rl   = lane & 15;
    const int kh   = hi16 * 8;     // half-index base within LDS row

    const int m0 = blockIdx.y * Bb;
    const int n0 = blockIdx.x * BN;

    v8f acc[4][2];
    #pragma unroll
    for (int i = 0; i < 4; ++i)
        #pragma unroll
        for (int j = 0; j < 2; ++j) acc[i][j] = (v8f){0.f,0.f,0.f,0.f,0.f,0.f,0.f,0.f};

    // each thread owns a 16-half strip of one tile row of A and of B
    const int lr = tid >> 1;            // 0..127 (tile row)
    const int lc = (tid & 1) * 16;      // 0 or 16 (half column)
    const _Float16* Ag = A  + (size_t)(m0 + lr) * K + lc;
    const _Float16* Bg = Bw + (size_t)(n0 + lr) * K + lc;

    int p = 0;

#if USE_ASYNC
    // preload tile 0 into buffer 0 (4 async b128 ops per thread/wave)
    ASYNC_CP16(Ag, &As[0][lr][lc], 0);
    ASYNC_CP16(Ag, &As[0][lr][lc], 16);   // imm offset advances global AND lds addr
    ASYNC_CP16(Bg, &Bs[0][lr][lc], 0);
    ASYNC_CP16(Bg, &Bs[0][lr][lc], 16);
#endif

    for (int k0 = 0; k0 < K; k0 += BKK) {
#if USE_ASYNC
        __syncthreads();   // everyone done reading the buffer we are about to overwrite
        if (k0 + BKK < K) {
            const _Float16* ga = Ag + k0 + BKK;
            const _Float16* gb = Bg + k0 + BKK;
            ASYNC_CP16(ga, &As[p ^ 1][lr][lc], 0);
            ASYNC_CP16(ga, &As[p ^ 1][lr][lc], 16);
            ASYNC_CP16(gb, &Bs[p ^ 1][lr][lc], 0);
            ASYNC_CP16(gb, &Bs[p ^ 1][lr][lc], 16);
            WAIT_ASYNC(4);   // async loads retire in order: current tile's 4 ops done
        } else {
            WAIT_ASYNC(0);
        }
        __syncthreads();   // current tile visible to all 8 waves
#else
        uint4 a0 = *(const uint4*)(Ag + k0);
        uint4 a1 = *(const uint4*)(Ag + k0 + 8);
        uint4 b0 = *(const uint4*)(Bg + k0);
        uint4 b1 = *(const uint4*)(Bg + k0 + 8);
        __syncthreads();
        *(uint4*)&As[0][lr][lc]     = a0;
        *(uint4*)&As[0][lr][lc + 8] = a1;
        *(uint4*)&Bs[0][lr][lc]     = b0;
        *(uint4*)&Bs[0][lr][lc + 8] = b1;
        __syncthreads();
#endif

        // fragments per ISA 16-bit layout: lanes 0-15 K 0-7/16-23, lanes 16-31 K 8-15/24-31
        v16h af[4], bf[2];
        #pragma unroll
        for (int i = 0; i < 4; ++i) {
            const int r = wm * 64 + i * 16 + rl;
            v8h lo = *(const v8h*)&As[p][r][kh];
            v8h hi = *(const v8h*)&As[p][r][kh + 16];
            af[i] = __builtin_shufflevector(lo, hi, 0,1,2,3,4,5,6,7,8,9,10,11,12,13,14,15);
        }
        #pragma unroll
        for (int j = 0; j < 2; ++j) {
            const int r = wn * 32 + j * 16 + rl;
            v8h lo = *(const v8h*)&Bs[p][r][kh];
            v8h hi = *(const v8h*)&Bs[p][r][kh + 16];
            bf[j] = __builtin_shufflevector(lo, hi, 0,1,2,3,4,5,6,7,8,9,10,11,12,13,14,15);
        }

        #pragma unroll
        for (int i = 0; i < 4; ++i)
            #pragma unroll
            for (int j = 0; j < 2; ++j)
                acc[i][j] = __builtin_amdgcn_wmma_f32_16x16x32_f16(
                    false, af[i], false, bf[j], (short)0, acc[i][j], false, false);

#if USE_ASYNC
        p ^= 1;
#endif
    }

    // epilogue; C/D layout: VGPR r -> row r (lanes<16) / r+8 (lanes>=16), col = lane&15
    #pragma unroll
    for (int i = 0; i < 4; ++i) {
        #pragma unroll
        for (int j = 0; j < 2; ++j) {
            const int col = n0 + wn * 32 + j * 16 + rl;
            const float bsv = bias[col];
            #pragma unroll
            for (int r = 0; r < 8; ++r) {
                const int row = m0 + wm * 64 + i * 16 + hi16 * 8 + r;
                float v = acc[i][j][r] + bsv;
                if (EPI == 0) {
                    float g = 0.5f * v * (1.0f + erff(v * 0.70710678118654752f));
                    outH[(size_t)row * N + col] = (_Float16)g;
                } else {
                    outF[(size_t)row * N + col] = v + Res[(size_t)row * N + col];
                }
            }
        }
    }
}

// ---------------- launch ----------------
extern "C" void kernel_launch(void* const* d_in, const int* in_sizes, int n_in,
                              void* d_out, int out_size, void* d_ws, size_t ws_size,
                              hipStream_t stream) {
    const float* x     = (const float*)d_in[0];
    const float* ln1_w = (const float*)d_in[1];
    const float* ln1_b = (const float*)d_in[2];
    const float* qkv_w = (const float*)d_in[3];
    const float* qkv_b = (const float*)d_in[4];
    const float* out_w = (const float*)d_in[5];
    const float* out_b = (const float*)d_in[6];
    const float* ln2_w = (const float*)d_in[7];
    const float* ln2_b = (const float*)d_in[8];
    const float* fc1_w = (const float*)d_in[9];
    const float* fc1_b = (const float*)d_in[10];
    const float* fc2_w = (const float*)d_in[11];
    const float* fc2_b = (const float*)d_in[12];
    float* out = (float*)d_out;

    char* ws = (char*)d_ws;
    size_t off = 0;
    float* mu      = (float*)(ws + off); off += (size_t)MROWS * 4;
    float* rstd    = (float*)(ws + off); off += (size_t)MROWS * 4;
    float* colmean = (float*)(ws + off); off += (size_t)BB * EE * 4;
    float* t1      = (float*)(ws + off); off += (size_t)BB * EE * 4;
    float* attnvec = (float*)(ws + off); off += (size_t)BB * EE * 4;
    float* x2      = (float*)(ws + off); off += (size_t)MROWS * EE * 4;
    _Float16* hln  = (_Float16*)(ws + off); off += (size_t)MROWS * EE * 2;
    _Float16* w1h  = (_Float16*)(ws + off); off += (size_t)FFD * EE * 2;
    _Float16* w2h  = (_Float16*)(ws + off); off += (size_t)EE * FFD * 2;
    _Float16* act  = (_Float16*)(ws + off); off += (size_t)MROWS * FFD * 2;

    // 1) LN1 row statistics
    k_ln_stats<<<MROWS, 256, 0, stream>>>(x, mu, rstd);

    // 2) column mean of LN1(x) over sequence (attention collapses to this)
    k_colmean<<<dim3(EE / 256, BB), 256, 0, stream>>>(x, mu, rstd, ln1_w, ln1_b, colmean);

    // 3) v-projection (rows [2E,3E) of qkv_w) then output projection
    k_gemv<<<dim3(EE / 256, BB), 256, 0, stream>>>(qkv_w + (size_t)2 * EE * EE,
                                                   qkv_b + 2 * EE, colmean, t1);
    k_gemv<<<dim3(EE / 256, BB), 256, 0, stream>>>(out_w, out_b, t1, attnvec);

    // 4) residual add + LN2 -> f16 activations
    k_add_ln2<<<MROWS, 256, 0, stream>>>(x, attnvec, ln2_w, ln2_b, x2, hln);

    // 5) weights to f16
    k_f32_to_f16<<<(FFD * EE + 255) / 256, 256, 0, stream>>>(fc1_w, w1h, FFD * EE);
    k_f32_to_f16<<<(EE * FFD + 255) / 256, 256, 0, stream>>>(fc2_w, w2h, EE * FFD);

    // 6) MLP GEMM1 (+bias, GELU) -> f16
    gemm_wmma<0><<<dim3(FFD / BN, MROWS / Bb), 256, 0, stream>>>(
        hln, w1h, fc1_b, nullptr, act, nullptr, MROWS, FFD, EE);

    // 7) MLP GEMM2 (+bias, +x2 residual) -> f32 output
    gemm_wmma<1><<<dim3(EE / BN, MROWS / Bb), 256, 0, stream>>>(
        act, w2h, fc2_b, x2, nullptr, out, MROWS, EE, FFD);
}
